// HybridQuantumClassicalModel_34634616275059
// MI455X (gfx1250) — matrix operand order repaired and checked by reading, not compiled, not run on hardware
//
#include <hip/hip_runtime.h>
#include <hip/hip_bf16.h>

typedef __attribute__((ext_vector_type(16))) _Float16 v16h;
typedef __attribute__((ext_vector_type(8)))  _Float16 v8h;
typedef __attribute__((ext_vector_type(8)))  float    v8f;

// exact pointee type from the compiler diagnostics:
//   param1: '__vector_size__(4*sizeof(int)) int __device__ *'  (global, AS1)
//   param2: '__vector_size__(4*sizeof(int)) int __shared__ *'  (LDS, AS3)
typedef int v4i_b128 __attribute__((__vector_size__(4 * sizeof(int))));

#define BATCH 16384

#if __has_builtin(__builtin_amdgcn_global_load_async_to_lds_b128)
#define HAVE_ASYNC_LDS 1
// offset applies to BOTH the global and LDS addresses (ISA §15.18 pseudocode)
#define ASYNC_CP16(dst, src, off)                                          \
    __builtin_amdgcn_global_load_async_to_lds_b128(                        \
        (__attribute__((address_space(1))) v4i_b128*)(src),                \
        (__attribute__((address_space(3))) v4i_b128*)(dst), (off), 0)
#else
#define HAVE_ASYNC_LDS 0
#endif

// ---------------------------------------------------------------------------
// f32 -> f16 elementwise convert
// ---------------------------------------------------------------------------
__global__ void convert_f16_kernel(const float* __restrict__ src,
                                   _Float16* __restrict__ dst, int n) {
    int i = blockIdx.x * blockDim.x + threadIdx.x;
    if (i < n) dst[i] = (_Float16)src[i];
}

// W [K,N] f32 row-major  ->  Wt [N,K] f16 row-major (i.e. column-major B)
__global__ void transpose_f16_kernel(const float* __restrict__ W,
                                     _Float16* __restrict__ Wt, int K, int N) {
    int i = blockIdx.x * blockDim.x + threadIdx.x;
    if (i >= K * N) return;
    int k = i / N, n = i - k * N;
    Wt[(size_t)n * K + k] = (_Float16)W[i];
}

// ---------------------------------------------------------------------------
// WMMA GEMM:  out = relu( A[M,K]f16 @ B[K,N] + bias )   (B given as [N,K] f16)
//   mode 0: store f16      mode 1: store f32
//   mode 2: also add rank-4 update qout[M,4] @ wq[4,N] (f32), store f32
// Block: 8 waves, 256 threads.  Block tile: 256(M) x 64(N).
// Wave tile: 32(M) x 64(N) = 8 accumulators.
// B panel staged in LDS (double-buffered KC=64 chunks) via async global->LDS
// copies shared by all 8 waves; A streamed from global (b128, coalesced).
// ---------------------------------------------------------------------------
#define KC   64
#define KCP  72   // padded row stride in f16 elems: 144B -> lane bases 4 dwords apart

__device__ __forceinline__ void stage_wait_barrier() {
#if HAVE_ASYNC_LDS
#if __has_builtin(__builtin_amdgcn_s_wait_asynccnt)
    __builtin_amdgcn_s_wait_asynccnt(0);
#else
    asm volatile("s_wait_asynccnt 0" ::: "memory");
#endif
#endif
    __syncthreads();
}

__global__ __launch_bounds__(256)
void gemm_wmma_kernel(const _Float16* __restrict__ A,
                      const _Float16* __restrict__ Bt,   // [N,K] f16
                      const float* __restrict__ bias,    // [N]
                      int K, int N, int mode,
                      const float* __restrict__ qout,    // [M,4] (mode 2)
                      const float* __restrict__ wq,      // [4,N] (mode 2)
                      _Float16* __restrict__ outH,       // mode 0
                      float* __restrict__ outF)          // mode 1/2
{
    __shared__ _Float16 Bsh[2][64 * KCP];

    const int tid     = threadIdx.x;
    const int lane    = tid & 31;
    const int wave    = tid >> 5;
    const int l15     = lane & 15;
    const int hi      = lane >> 4;                 // 0: lanes 0-15, 1: lanes 16-31
    const int rowBase = (blockIdx.x * 8 + wave) * 32;
    const int colBase = blockIdx.y * 64;

    // staging assignment: thread -> (column cc, 16 K-elems at ktb) of the panel
    const int cc  = tid >> 2;          // 0..63
    const int ktb = (tid & 3) * 16;    // 0,16,32,48
    const _Float16* stageSrcBase = Bt + (size_t)(colBase + cc) * K + ktb;

    // A row pointers for the two 16-row fragments (A-frag: hi half takes K+8)
    const _Float16* aRow0 = A + (size_t)(rowBase + l15) * K + (hi ? 8 : 0);
    const _Float16* aRow1 = aRow0 + (size_t)16 * K;

    v8f acc[2][4];
#pragma unroll
    for (int m = 0; m < 2; ++m)
#pragma unroll
        for (int t = 0; t < 4; ++t)
#pragma unroll
            for (int v = 0; v < 8; ++v) acc[m][t][v] = 0.0f;

    const int nchunks = K / KC;

    // ---- stage chunk 0 into buffer 0 ----
    {
        const _Float16* src = stageSrcBase;
        _Float16* dst = &Bsh[0][cc * KCP + ktb];
#if HAVE_ASYNC_LDS
        ASYNC_CP16(dst, src, 0);
        ASYNC_CP16(dst, src, 16);
#else
        *(v8h*)dst       = *(const v8h*)src;
        *(v8h*)(dst + 8) = *(const v8h*)(src + 8);
#endif
    }

    for (int c = 0; c < nchunks; ++c) {
        stage_wait_barrier();                 // chunk c resident; prev reads done

        if (c + 1 < nchunks) {                // prefetch chunk c+1 into other buf
            const _Float16* src = stageSrcBase + (c + 1) * KC;
            _Float16* dst = &Bsh[(c + 1) & 1][cc * KCP + ktb];
#if HAVE_ASYNC_LDS
            ASYNC_CP16(dst, src, 0);
            ASYNC_CP16(dst, src, 16);
#else
            *(v8h*)dst       = *(const v8h*)src;
            *(v8h*)(dst + 8) = *(const v8h*)(src + 8);
#endif
        }

        const _Float16* bp = &Bsh[c & 1][0];
        const int kk = c * KC;

#pragma unroll
        for (int ks = 0; ks < 2; ++ks) {      // two 16x16x32 K-steps per chunk
            const int ko = kk + ks * 32;

            v16h a0, a1;
            {
                const v8h a0lo = *(const v8h*)(aRow0 + ko);
                const v8h a0hi = *(const v8h*)(aRow0 + ko + 16);
                const v8h a1lo = *(const v8h*)(aRow1 + ko);
                const v8h a1hi = *(const v8h*)(aRow1 + ko + 16);
#pragma unroll
                for (int i = 0; i < 8; ++i) {
                    a0[i] = a0lo[i]; a0[8 + i] = a0hi[i];
                    a1[i] = a1lo[i]; a1[8 + i] = a1hi[i];
                }
            }
#pragma unroll
            for (int t = 0; t < 4; ++t) {
                // B frag: col = colBase + t*16 + l15, K = ks*32 + hi*8 + {0..7, 16..23}
                const _Float16* bf =
                    bp + (t * 16 + l15) * KCP + ks * 32 + (hi ? 8 : 0);
                v16h b;
                {
                    const v8h blo = *(const v8h*)(bf);
                    const v8h bhi = *(const v8h*)(bf + 16);
#pragma unroll
                    for (int i = 0; i < 8; ++i) { b[i] = blo[i]; b[8 + i] = bhi[i]; }
                }
                acc[0][t] = __builtin_amdgcn_wmma_f32_16x16x32_f16(
                    false, a0, false, b, (short)0, acc[0][t], false, false);
                acc[1][t] = __builtin_amdgcn_wmma_f32_16x16x32_f16(
                    false, a1, false, b, (short)0, acc[1][t], false, false);
            }
        }
    }

    // epilogue: bias (+ rank-4 qout update) + ReLU + store
#pragma unroll
    for (int m = 0; m < 2; ++m) {
#pragma unroll
        for (int t = 0; t < 4; ++t) {
            const int col = colBase + t * 16 + l15;
            const float bv = bias[col];
#pragma unroll
            for (int v = 0; v < 8; ++v) {
                const int r = rowBase + m * 16 + v + (hi ? 8 : 0);
                float val = acc[m][t][v] + bv;
                if (mode == 2) {
                    const float* q = qout + (size_t)r * 4;
                    val += q[0] * wq[col] + q[1] * wq[N + col] +
                           q[2] * wq[2 * N + col] + q[3] * wq[3 * N + col];
                }
                val = fmaxf(val, 0.0f);
                const size_t o = (size_t)r * N + col;
                if (mode == 0) outH[o] = (_Float16)val;
                else           outF[o] = val;
            }
        }
    }
}

// ---------------------------------------------------------------------------
// 4-qubit statevector sim per sample (fully unrolled, 16 complex amps in regs)
// ---------------------------------------------------------------------------
__global__ void qsim_kernel(const float* __restrict__ h2,   // [B,64]
                            const float* __restrict__ w3,   // [64,4]
                            const float* __restrict__ b3,   // [4]
                            const float* __restrict__ qp,   // [2,4]
                            float* __restrict__ qout)       // [B,4]
{
    const int r = blockIdx.x * blockDim.x + threadIdx.x;
    if (r >= BATCH) return;

    const float* hrow = h2 + (size_t)r * 64;
    float ang[4] = {b3[0], b3[1], b3[2], b3[3]};
    for (int k = 0; k < 64; ++k) {
        const float hv = hrow[k];
        const float4 w = *(const float4*)(w3 + k * 4);
        ang[0] += hv * w.x; ang[1] += hv * w.y;
        ang[2] += hv * w.z; ang[3] += hv * w.w;
    }
#pragma unroll
    for (int j = 0; j < 4; ++j) ang[j] = tanhf(ang[j]);

    float re[16], im[16];
#pragma unroll
    for (int i = 0; i < 16; ++i) { re[i] = 0.25f; im[i] = 0.0f; }

#pragma unroll
    for (int w = 0; w < 4; ++w) {
        const float h = 0.5f * ang[w];
        const float c = cosf(h), s = sinf(h);
#pragma unroll
        for (int i = 0; i < 16; ++i) {
            const float sg = ((i >> (3 - w)) & 1) ? s : -s;
            const float nr = re[i] * c - im[i] * sg;
            const float ni = re[i] * sg + im[i] * c;
            re[i] = nr; im[i] = ni;
        }
    }

#define CNOT(cq, tq)                                                          \
    {                                                                          \
        const int cm = 1 << (3 - (cq)), tm = 1 << (3 - (tq));                  \
        _Pragma("unroll")                                                      \
        for (int i = 0; i < 16; ++i)                                           \
            if ((i & cm) && !(i & tm)) {                                       \
                const int j = i | tm;                                          \
                float tr = re[i]; re[i] = re[j]; re[j] = tr;                   \
                float ti = im[i]; im[i] = im[j]; im[j] = ti;                   \
            }                                                                  \
    }

    CNOT(0, 1); CNOT(1, 2); CNOT(2, 3);

#pragma unroll
    for (int layer = 0; layer < 2; ++layer) {
#pragma unroll
        for (int w = 0; w < 4; ++w) {
            const float h = 0.5f * qp[layer * 4 + w];
            const float c = cosf(h), s = sinf(h);
            const int m = 1 << (3 - w);
#pragma unroll
            for (int i = 0; i < 16; ++i)
                if (!(i & m)) {
                    const int j = i | m;
                    const float r0 = re[i], i0 = im[i];
                    const float r1 = re[j], i1 = im[j];
                    re[i] = c * r0 - s * r1; im[i] = c * i0 - s * i1;
                    re[j] = s * r0 + c * r1; im[j] = s * i0 + c * i1;
                }
        }
        CNOT(0, 1); CNOT(1, 2); CNOT(2, 3); CNOT(3, 0);
    }
#undef CNOT

    float z[4] = {0.f, 0.f, 0.f, 0.f};
#pragma unroll
    for (int i = 0; i < 16; ++i) {
        const float p = re[i] * re[i] + im[i] * im[i];
#pragma unroll
        for (int w = 0; w < 4; ++w)
            z[w] += ((i >> (3 - w)) & 1) ? -p : p;
    }
    float4* o = (float4*)(qout + (size_t)r * 4);
    *o = make_float4(z[0], z[1], z[2], z[3]);
}

// ---------------------------------------------------------------------------
// BatchNorm column stats: mean + rsqrt(biased var + eps) per column
// ---------------------------------------------------------------------------
__global__ void bn_stats_kernel(const float* __restrict__ Z, int N,
                                float* __restrict__ mean,
                                float* __restrict__ rstd) {
    const int col = blockIdx.x;
    float s = 0.f, s2 = 0.f;
    for (int r = threadIdx.x; r < BATCH; r += 256) {
        const float v = Z[(size_t)r * N + col];
        s += v; s2 += v * v;
    }
    __shared__ float sh[256], sh2[256];
    sh[threadIdx.x] = s; sh2[threadIdx.x] = s2;
    __syncthreads();
    for (int st = 128; st > 0; st >>= 1) {
        if (threadIdx.x < st) {
            sh[threadIdx.x]  += sh[threadIdx.x + st];
            sh2[threadIdx.x] += sh2[threadIdx.x + st];
        }
        __syncthreads();
    }
    if (threadIdx.x == 0) {
        const float m = sh[0] * (1.0f / BATCH);
        const float v = sh2[0] * (1.0f / BATCH) - m * m;
        mean[col] = m;
        rstd[col] = rsqrtf(v + 1e-5f);
    }
}

__global__ void bn_apply_kernel(const float* __restrict__ Z,
                                const float* __restrict__ mean,
                                const float* __restrict__ rstd,
                                const float* __restrict__ g,
                                const float* __restrict__ be,
                                _Float16* __restrict__ out, int N) {
    const long i = (long)blockIdx.x * blockDim.x + threadIdx.x;
    if (i >= (long)BATCH * N) return;
    const int n = (int)(i % N);
    out[i] = (_Float16)((Z[i] - mean[n]) * rstd[n] * g[n] + be[n]);
}

// ---------------------------------------------------------------------------
// final tiny GEMM: out[B,4] = h[B,256]f16 @ wc3[256,4] + bc3
// ---------------------------------------------------------------------------
__global__ void final_kernel(const _Float16* __restrict__ h,
                             const float* __restrict__ wc3,
                             const float* __restrict__ bc3,
                             float* __restrict__ out) {
    const int r = blockIdx.x * blockDim.x + threadIdx.x;
    if (r >= BATCH) return;
    float a0 = bc3[0], a1 = bc3[1], a2 = bc3[2], a3 = bc3[3];
    const _Float16* hr = h + (size_t)r * 256;
    for (int k = 0; k < 256; ++k) {
        const float hv = (float)hr[k];
        const float4 w = *(const float4*)(wc3 + k * 4);
        a0 += hv * w.x; a1 += hv * w.y; a2 += hv * w.z; a3 += hv * w.w;
    }
    float4* o = (float4*)(out + (size_t)r * 4);
    *o = make_float4(a0, a1, a2, a3);
}

// ---------------------------------------------------------------------------
// workspace layout (bytes)
// ---------------------------------------------------------------------------
#define OFF_XH    ((size_t)0)                               // [B,1024] f16
#define OFF_W1T   (OFF_XH   + (size_t)BATCH * 1024 * 2)     // [256,1024] f16
#define OFF_W2T   (OFF_W1T  + (size_t)256 * 1024 * 2)       // [64,256]  f16
#define OFF_WC1T  (OFF_W2T  + (size_t)64 * 256 * 2)         // [512,1024] f16
#define OFF_WC2T  (OFF_WC1T + (size_t)512 * 1024 * 2)       // [256,512] f16
#define OFF_H1    (OFF_WC2T + (size_t)256 * 512 * 2)        // [B,256] f16
#define OFF_H2    (OFF_H1   + (size_t)BATCH * 256 * 2)      // [B,64]  f32
#define OFF_QOUT  (OFF_H2   + (size_t)BATCH * 64 * 4)       // [B,4]   f32
#define OFF_Z1    (OFF_QOUT + (size_t)BATCH * 4 * 4)        // [B,512] f32
#define OFF_H1C   (OFF_Z1   + (size_t)BATCH * 512 * 4)      // [B,512] f16
#define OFF_Z2    (OFF_H1C  + (size_t)BATCH * 512 * 2)      // [B,256] f32
#define OFF_H2C   (OFF_Z2   + (size_t)BATCH * 256 * 4)      // [B,256] f16
#define OFF_MEAN1 (OFF_H2C  + (size_t)BATCH * 256 * 2)
#define OFF_RSTD1 (OFF_MEAN1 + 512 * 4)
#define OFF_MEAN2 (OFF_RSTD1 + 512 * 4)
#define OFF_RSTD2 (OFF_MEAN2 + 256 * 4)

extern "C" void kernel_launch(void* const* d_in, const int* in_sizes, int n_in,
                              void* d_out, int out_size, void* d_ws, size_t ws_size,
                              hipStream_t stream) {
    (void)in_sizes; (void)n_in; (void)out_size; (void)ws_size;

    const float* features = (const float*)d_in[0];
    const float* w1  = (const float*)d_in[1];
    const float* b1  = (const float*)d_in[2];
    const float* w2  = (const float*)d_in[3];
    const float* b2  = (const float*)d_in[4];
    const float* w3  = (const float*)d_in[5];
    const float* b3  = (const float*)d_in[6];
    const float* qp  = (const float*)d_in[7];
    const float* wc1 = (const float*)d_in[8];
    const float* bc1 = (const float*)d_in[9];
    const float* g1  = (const float*)d_in[10];
    const float* be1 = (const float*)d_in[11];
    const float* wc2 = (const float*)d_in[12];
    const float* bc2 = (const float*)d_in[13];
    const float* g2  = (const float*)d_in[14];
    const float* be2 = (const float*)d_in[15];
    const float* wc3 = (const float*)d_in[16];
    const float* bc3 = (const float*)d_in[17];
    float* out = (float*)d_out;

    char* ws = (char*)d_ws;
    _Float16* Xh   = (_Float16*)(ws + OFF_XH);
    _Float16* W1t  = (_Float16*)(ws + OFF_W1T);
    _Float16* W2t  = (_Float16*)(ws + OFF_W2T);
    _Float16* Wc1t = (_Float16*)(ws + OFF_WC1T);
    _Float16* Wc2t = (_Float16*)(ws + OFF_WC2T);
    _Float16* H1   = (_Float16*)(ws + OFF_H1);
    float*    H2   = (float*)(ws + OFF_H2);
    float*    QOUT = (float*)(ws + OFF_QOUT);
    float*    Z1   = (float*)(ws + OFF_Z1);
    _Float16* H1C  = (_Float16*)(ws + OFF_H1C);
    float*    Z2   = (float*)(ws + OFF_Z2);
    _Float16* H2C  = (_Float16*)(ws + OFF_H2C);
    float*    M1   = (float*)(ws + OFF_MEAN1);
    float*    R1   = (float*)(ws + OFF_RSTD1);
    float*    M2   = (float*)(ws + OFF_MEAN2);
    float*    R2   = (float*)(ws + OFF_RSTD2);

    const dim3 blk(256);

    // --- precision conversion / weight transposes ---
    convert_f16_kernel<<<(BATCH * 1024) / 256, blk, 0, stream>>>(features, Xh, BATCH * 1024);
    transpose_f16_kernel<<<(1024 * 256) / 256, blk, 0, stream>>>(w1, W1t, 1024, 256);
    transpose_f16_kernel<<<(256 * 64) / 256, blk, 0, stream>>>(w2, W2t, 256, 64);
    transpose_f16_kernel<<<(1024 * 512) / 256, blk, 0, stream>>>(wc1 + 4 * 512, Wc1t, 1024, 512);
    transpose_f16_kernel<<<(512 * 256) / 256, blk, 0, stream>>>(wc2, Wc2t, 512, 256);

    // --- reducer GEMM 1: [B,1024] @ [1024,256] -> relu -> f16 ---
    gemm_wmma_kernel<<<dim3(BATCH / 256, 256 / 64), blk, 0, stream>>>(
        Xh, W1t, b1, 1024, 256, 0, nullptr, nullptr, H1, nullptr);

    // --- reducer GEMM 2: [B,256] @ [256,64] -> relu -> f32 ---
    gemm_wmma_kernel<<<dim3(BATCH / 256, 64 / 64), blk, 0, stream>>>(
        H1, W2t, b2, 256, 64, 1, nullptr, nullptr, nullptr, H2);

    // --- quantum layer: angles = tanh(H2 @ w3 + b3); 4-qubit sim ---
    qsim_kernel<<<BATCH / 256, blk, 0, stream>>>(H2, w3, b3, qp, QOUT);

    // --- classifier GEMM 1: [qout|features] @ wc1 -> relu -> f32 ---
    gemm_wmma_kernel<<<dim3(BATCH / 256, 512 / 64), blk, 0, stream>>>(
        Xh, Wc1t, bc1, 1024, 512, 2, QOUT, wc1, nullptr, Z1);
    bn_stats_kernel<<<512, blk, 0, stream>>>(Z1, 512, M1, R1);
    bn_apply_kernel<<<(BATCH * 512) / 256, blk, 0, stream>>>(Z1, M1, R1, g1, be1, H1C, 512);

    // --- classifier GEMM 2: [B,512] @ [512,256] -> relu -> BN -> f16 ---
    gemm_wmma_kernel<<<dim3(BATCH / 256, 256 / 64), blk, 0, stream>>>(
        H1C, Wc2t, bc2, 512, 256, 1, nullptr, nullptr, nullptr, Z2);
    bn_stats_kernel<<<256, blk, 0, stream>>>(Z2, 256, M2, R2);
    bn_apply_kernel<<<(BATCH * 256) / 256, blk, 0, stream>>>(Z2, M2, R2, g2, be2, H2C, 256);

    // --- final logits ---
    final_kernel<<<BATCH / 256, blk, 0, stream>>>(H2C, wc3, bc3, out);
}
